// GIN_36713380446845
// MI455X (gfx1250) — matrix operand order repaired and checked by reading, not compiled, not run on hardware
//
#include <hip/hip_runtime.h>
#include <hip/hip_bf16.h>

// ---------------------------------------------------------------------------
// CDNA5 (gfx1250) GIN forward. wave32.
//  - GEMMs: v_wmma_f32_16x16x32_bf16; weights LDS-swizzled to fragment order
//    (B frag = 2x ds_load_b128); A panels double-buffered in LDS via
//    global_load_async_to_lds_b128 with s_wait_asynccnt 8 overlap (ASYNCcnt
//    in-order completion => pipelined copy/compute).
//  - bf16 packing done pairwise into u32 (no v_mov_b16 chains).
//  - Edge aggregation: wave-per-edge f32 global atomics (bandwidth bound).
// ---------------------------------------------------------------------------

typedef __attribute__((ext_vector_type(16))) __bf16   v16bf;
typedef __attribute__((ext_vector_type(8)))  __bf16   v8bf;
typedef __attribute__((ext_vector_type(8)))  float    v8f;
typedef __attribute__((ext_vector_type(8)))  unsigned v8ui;

__device__ __forceinline__ unsigned short f32_to_bf16_bits(float f) {
    unsigned int u = __builtin_bit_cast(unsigned int, f);
    return (unsigned short)((u + 0x7FFFu + ((u >> 16) & 1u)) >> 16); // RNE
}
// Two f32 -> packed {lo,hi} bf16 in one u32 (RNE), avoids 16-bit mov packing.
__device__ __forceinline__ unsigned pack2_bf16(float f0, float f1) {
    unsigned u0 = __builtin_bit_cast(unsigned, f0);
    unsigned u1 = __builtin_bit_cast(unsigned, f1);
    u0 = (u0 + 0x7FFFu + ((u0 >> 16) & 1u)) >> 16;
    u1 = (u1 + 0x7FFFu + ((u1 >> 16) & 1u)) & 0xFFFF0000u;
    return u0 | u1;
}
__device__ __forceinline__ v16bf combine16(v8bf lo, v8bf hi) {
    return __builtin_shufflevector(lo, hi,
        0, 1, 2, 3, 4, 5, 6, 7, 8, 9, 10, 11, 12, 13, 14, 15);
}

// Async global->LDS copy, 16B per lane. VDST = per-lane LDS byte address,
// VADDR = per-lane 64-bit global address. Tracked by ASYNCcnt.
__device__ __forceinline__ void async_g2l_b128(unsigned lds_off, const void* g) {
    asm volatile("global_load_async_to_lds_b128 %0, %1, off"
                 :: "v"(lds_off), "v"(g) : "memory");
}
template <int CNT>
__device__ __forceinline__ void wait_async_le() {
    asm volatile("s_wait_asynccnt %0" :: "i"(CNT) : "memory");
}
__device__ __forceinline__ unsigned lds_addr32(const void* p) {
    // generic pointer to LDS: low 32 bits are the LDS byte address
    return (unsigned)(unsigned long long)p;
}

// --------------------------- utility kernels -------------------------------

__global__ void cvt_f32_bf16(const float* __restrict__ src,
                             unsigned short* __restrict__ dst, int n) {
    int i = blockIdx.x * blockDim.x + threadIdx.x;
    if (i < n) dst[i] = f32_to_bf16_bits(src[i]);
}

__global__ void copy_f32x4(const float* __restrict__ src,
                           float* __restrict__ dst, long long n4) {
    long long i = (long long)blockIdx.x * blockDim.x + threadIdx.x;
    if (i < n4) ((float4*)dst)[i] = ((const float4*)src)[i];
}

// One wave per edge: gather 128 floats from src row, atomic-add into dst row.
__global__ void edge_agg(const float* __restrict__ X, float* __restrict__ OUT,
                         const long long* __restrict__ src,
                         const long long* __restrict__ dst, int nEdges) {
    int e    = (int)(((long long)blockIdx.x * blockDim.x + threadIdx.x) >> 5);
    int lane = threadIdx.x & 31;
    if (e >= nEdges) return;
    long long s = src[e];
    long long d = dst[e];
    float4 v = *(const float4*)(X + s * 128 + lane * 4);
    float* o = OUT + d * 128 + lane * 4;
    atomicAdd(o + 0, v.x);
    atomicAdd(o + 1, v.y);
    atomicAdd(o + 2, v.z);
    atomicAdd(o + 3, v.w);
}

// One wave per row: L2 normalize 128 floats (wave32 shuffle reduction).
__global__ void l2norm_rows(const float* __restrict__ X, float* __restrict__ Y,
                            int nRows) {
    int r    = (int)(((long long)blockIdx.x * blockDim.x + threadIdx.x) >> 5);
    int lane = threadIdx.x & 31;
    if (r >= nRows) return;
    float4 v = *(const float4*)(X + (size_t)r * 128 + lane * 4);
    float ss = v.x * v.x + v.y * v.y + v.z * v.z + v.w * v.w;
    #pragma unroll
    for (int off = 16; off > 0; off >>= 1) ss += __shfl_xor(ss, off, 32);
    float inv = 1.0f / fmaxf(sqrtf(ss), 1e-12f);
    float4 o = {v.x * inv, v.y * inv, v.z * inv, v.w * inv};
    *(float4*)(Y + (size_t)r * 128 + lane * 4) = o;
}

// --------------------------- WMMA GEMM -------------------------------------
// D[M,N] = relu(A[M,K](f32) @ W[K,N](bf16) + bias[N]).
// Block = 128 thr = 4 waves; wave owns a 16-row tile and all N/16 col tiles.
// K processed in KP=64 panels: weight panel LDS-swizzled into B-fragment
// order; the wave's 16xKP A panel is double-buffered in LDS via async copies
// pipelined one panel ahead (s_wait_asynccnt KP/8 keeps only the in-flight
// prefetch outstanding).
template <int K, int N>
__global__ void gemm_bias_relu(const float* __restrict__ A,
                               const unsigned short* __restrict__ W,
                               const float* __restrict__ bias,
                               float* __restrict__ D, int M) {
    constexpr int NT  = N / 16;               // column tiles
    constexpr int KP  = (K >= 64) ? 64 : K;   // k-panel size
    constexpr int KC2 = KP / 32;              // chunks per panel
    constexpr int NP  = K / KP;               // panels
    constexpr int NW  = 4;                    // waves per block

    __shared__ __align__(16) unsigned short wpanel[KC2 * NT * 32 * 16]; // KP*N
    __shared__ __align__(16) float atile[NW][2][16 * KP];

    const int wave = threadIdx.x >> 5;
    const int lane = threadIdx.x & 31;
    const int m0   = (blockIdx.x * NW + wave) * 16;
    const int l15  = lane & 15;
    const int kh   = lane >> 4;
    const char* Abytes = (const char*)A;

    // Issue the (16 x KP) A panel copy for panel p into LDS buffer `buf`.
    auto issueA = [&](int p, int buf) {
        const unsigned base = lds_addr32(&atile[wave][buf][0]);
        #pragma unroll
        for (int i = 0; i < KP / 8; ++i) {        // (16*KP*4)/512 instrs
            int linear = i * 512 + lane * 16;     // byte index in tile
            int r  = linear / (KP * 4);
            int cb = linear - r * (KP * 4);
            const void* g = Abytes + ((size_t)(m0 + r) * K + p * KP) * 4 + cb;
            async_g2l_b128(base + (unsigned)linear, g);
        }
    };

    issueA(0, 0);
    v8f acc[NT] = {};

    for (int p = 0; p < NP; ++p) {
        __syncthreads();   // previous weight panel fully consumed

        // ---- stage W panel, swizzled to B-fragment order ------------------
        // frag element j of lane (kh2*16+ln), tile t, chunk kc2 is
        // W[p*KP + kc2*32 + (j<8 ? kh2*8+j : 16+kh2*8+j-8)][t*16 + ln]
        for (int i = threadIdx.x; i < KP * N; i += NW * 32) {
            int k = i / N, n = i - (i / N) * N;
            int kc2 = k >> 5, kh2 = (k >> 3) & 1, j = (k & 7) + ((k >> 4) << 3);
            int t = n >> 4, ln = n & 15;
            wpanel[(((kc2 * NT + t) * 32) + kh2 * 16 + ln) * 16 + j] =
                W[(size_t)(p * KP + k) * N + n];
        }
        if (p + 1 < NP) {
            __builtin_prefetch(W + (size_t)(p + 1) * KP * N + threadIdx.x * 8, 0, 0);
            issueA(p + 1, (p + 1) & 1);           // prefetch next A panel
            wait_async_le<KP / 8>();              // panel p complete (in order)
        } else {
            wait_async_le<0>();
        }
        __syncthreads();    // wpanel visible to all waves

        // ---- compute ------------------------------------------------------
        const float* at = &atile[wave][p & 1][0];
        for (int kc2 = 0; kc2 < KC2; ++kc2) {
            // A fragment: lanes 0-15 hold K 0..7 & 16..23 of the chunk,
            // lanes 16-31 hold K 8..15 & 24..31 (two 32B LDS reads)
            const float* ar = at + l15 * KP + kc2 * 32 + kh * 8;
            v8ui au;
            #pragma unroll
            for (int jj = 0; jj < 4; ++jj)
                au[jj]     = pack2_bf16(ar[2 * jj],      ar[2 * jj + 1]);
            #pragma unroll
            for (int jj = 0; jj < 4; ++jj)
                au[4 + jj] = pack2_bf16(ar[16 + 2 * jj], ar[17 + 2 * jj]);
            v16bf a = __builtin_bit_cast(v16bf, au);

            #pragma unroll
            for (int t = 0; t < NT; ++t) {
                const v8bf* bp =
                    (const v8bf*)&wpanel[((kc2 * NT + t) * 32 + lane) * 16];
                v16bf b = combine16(bp[0], bp[1]);
                acc[t] = __builtin_amdgcn_wmma_f32_16x16x32_bf16(
                    false, a, false, b, (short)0, acc[t], false, false);
            }
        }
    }

    // ---- bias + relu + store ---------------------------------------------
    #pragma unroll
    for (int t = 0; t < NT; ++t) {
        const int n  = t * 16 + l15;
        const float bn = bias[n];
        #pragma unroll
        for (int r = 0; r < 8; ++r) {
            const int m = m0 + r + 8 * kh;        // C/D layout rows r / r+8
            float v = acc[t][r] + bn;
            D[(size_t)m * N + n] = v > 0.f ? v : 0.f;
        }
    }
}

// --------------------------- SOPOOL ----------------------------------------
// One block (128 thr = 4 waves) per graph: HH[g] = sg^T sg, sg = [256,32].
// sg staged in LDS pre-swizzled into BOTH WMMA A-fragment and B-fragment
// order, so every fragment is two contiguous ds_load_b128.
__global__ void sopool(const float* __restrict__ S, float* __restrict__ HH) {
    __shared__ __align__(16) unsigned short lda[8 * 2 * 32 * 16]; // A-frag order
    __shared__ __align__(16) unsigned short ldb[8 * 2 * 32 * 16]; // B-frag order
    const int g = blockIdx.x;
    const float* sg = S + (size_t)g * 256 * 32;

    for (int i = threadIdx.x; i < 256 * 32; i += 128) {
        int k = i >> 5, f = i & 31;               // sg[k][f]
        unsigned short v = f32_to_bf16_bits(sg[i]);
        int kc = k >> 5, r = k & 31;
        // B fragment: element j of lane (khB*16+n15) is B[kc*32+khB*16+j][n]
        int khB = (r >> 4) & 1, jB = r & 15;
        ldb[(((kc * 2 + (f >> 4)) * 32) + khB * 16 + (f & 15)) * 16 + jB] = v;
        // A fragment: element j of lane (khA*16+m) is A[m][kc*32+khA*16+j]
        int khA = (r >> 4) & 1, jA = r & 15;
        lda[(((kc * 2 + (f >> 4)) * 32) + khA * 16 + (f & 15)) * 16 + jA] = v;
    }
    __syncthreads();

    const int wave = threadIdx.x >> 5;
    const int lane = threadIdx.x & 31;
    const int mt = wave >> 1, nt = wave & 1;
    const int l15 = lane & 15, kh = lane >> 4;

    v8f acc = {};
    for (int kc = 0; kc < 8; ++kc) {
        const v8bf* ap = (const v8bf*)&lda[((kc * 2 + mt) * 32 + lane) * 16];
        const v8bf* bp = (const v8bf*)&ldb[((kc * 2 + nt) * 32 + lane) * 16];
        v16bf a = combine16(ap[0], ap[1]);
        v16bf b = combine16(bp[0], bp[1]);
        acc = __builtin_amdgcn_wmma_f32_16x16x32_bf16(
            false, a, false, b, (short)0, acc, false, false);
    }
    #pragma unroll
    for (int r = 0; r < 8; ++r) {
        const int m = mt * 16 + r + 8 * kh;
        const int n = nt * 16 + l15;
        HH[(size_t)g * 1024 + m * 32 + n] = acc[r];
    }
}

// Final tiny layer: out[G,2] = relu(O2[G,32] @ Wm3[32,2] + bm3)
__global__ void final_fc(const float* __restrict__ O2,
                         const float* __restrict__ Wm3,
                         const float* __restrict__ bm3,
                         float* __restrict__ OUT, int G) {
    int i = blockIdx.x * blockDim.x + threadIdx.x;
    if (i >= G * 2) return;
    int g = i >> 1, c = i & 1;
    float acc = bm3[c];
    #pragma unroll
    for (int k = 0; k < 32; ++k) acc += O2[g * 32 + k] * Wm3[k * 2 + c];
    OUT[g * 2 + c] = acc > 0.f ? acc : 0.f;
}

// --------------------------- launch ----------------------------------------

extern "C" void kernel_launch(void* const* d_in, const int* in_sizes, int n_in,
                              void* d_out, int out_size, void* d_ws, size_t ws_size,
                              hipStream_t stream) {
    (void)n_in; (void)out_size; (void)ws_size;

    const float*     x   = (const float*)d_in[0];
    const long long* ei  = (const long long*)d_in[1];   // int64 per reference
    const float* W1a = (const float*)d_in[2];  const float* b1a = (const float*)d_in[3];
    const float* W2a = (const float*)d_in[4];  const float* b2a = (const float*)d_in[5];
    const float* W1b = (const float*)d_in[6];  const float* b1b = (const float*)d_in[7];
    const float* W2b = (const float*)d_in[8];  const float* b2b = (const float*)d_in[9];
    const float* Ws1 = (const float*)d_in[10]; const float* bs1 = (const float*)d_in[11];
    const float* Ws2 = (const float*)d_in[12]; const float* bs2 = (const float*)d_in[13];
    const float* Ws3 = (const float*)d_in[14]; const float* bs3 = (const float*)d_in[15];
    const float* Wm1 = (const float*)d_in[16]; const float* bm1 = (const float*)d_in[17];
    const float* Wm2 = (const float*)d_in[18]; const float* bm2 = (const float*)d_in[19];
    const float* Wm3 = (const float*)d_in[20]; const float* bm3 = (const float*)d_in[21];

    const int Nn = in_sizes[0] / 128;   // 131072 nodes
    const int E  = in_sizes[1] / 2;     // 2097152 edges
    const int G  = Nn / 256;            // 512 graphs

    // ---- workspace layout ----
    float* bufA = (float*)d_ws;                       // [Nn,128] f32
    float* bufB = bufA + (size_t)Nn * 128;            // [Nn,128] f32
    unsigned short* wb  = (unsigned short*)(bufB + (size_t)Nn * 128);
    unsigned short* w1a = wb;                 // 128*128
    unsigned short* w2a = w1a + 128 * 128;
    unsigned short* w1b = w2a + 128 * 128;
    unsigned short* w2b = w1b + 128 * 128;
    unsigned short* ws1 = w2b + 128 * 128;    // 128*32
    unsigned short* ws2 = ws1 + 128 * 32;     // 32*32
    unsigned short* ws3 = ws2 + 32 * 32;      // 32*32
    unsigned short* wm1 = ws3 + 32 * 32;      // 1024*32
    unsigned short* wm2 = wm1 + 1024 * 32;    // 32*32

    // ---- weight conversion f32 -> bf16 ----
    cvt_f32_bf16<<<dim3((128*128 + 255)/256), dim3(256), 0, stream>>>(W1a, w1a, 128*128);
    cvt_f32_bf16<<<dim3((128*128 + 255)/256), dim3(256), 0, stream>>>(W2a, w2a, 128*128);
    cvt_f32_bf16<<<dim3((128*128 + 255)/256), dim3(256), 0, stream>>>(W1b, w1b, 128*128);
    cvt_f32_bf16<<<dim3((128*128 + 255)/256), dim3(256), 0, stream>>>(W2b, w2b, 128*128);
    cvt_f32_bf16<<<dim3((128*32  + 255)/256), dim3(256), 0, stream>>>(Ws1, ws1, 128*32);
    cvt_f32_bf16<<<dim3((32*32   + 255)/256), dim3(256), 0, stream>>>(Ws2, ws2, 32*32);
    cvt_f32_bf16<<<dim3((32*32   + 255)/256), dim3(256), 0, stream>>>(Ws3, ws3, 32*32);
    cvt_f32_bf16<<<dim3((1024*32 + 255)/256), dim3(256), 0, stream>>>(Wm1, wm1, 1024*32);
    cvt_f32_bf16<<<dim3((32*32   + 255)/256), dim3(256), 0, stream>>>(Wm2, wm2, 32*32);

    const long long n4 = (long long)Nn * 128 / 4;
    const dim3 cpGrid((unsigned)((n4 + 255) / 256));
    const dim3 aggGrid((unsigned)(((long long)E * 32 + 255) / 256));
    const dim3 gemmGridN((unsigned)(Nn / 64));    // 4 waves * 16 rows / block
    const dim3 gemmGridG((unsigned)(G / 64));

    // ---- GIN layer 1 ----
    copy_f32x4<<<cpGrid, 256, 0, stream>>>(x, bufA, n4);
    edge_agg<<<aggGrid, 256, 0, stream>>>(x, bufA, ei, ei + E, E);
    gemm_bias_relu<128,128><<<gemmGridN, 128, 0, stream>>>(bufA, w1a, b1a, bufB, Nn);
    gemm_bias_relu<128,128><<<gemmGridN, 128, 0, stream>>>(bufB, w2a, b2a, bufA, Nn);

    // ---- GIN layer 2 (h1 in bufA) ----
    copy_f32x4<<<cpGrid, 256, 0, stream>>>(bufA, bufB, n4);
    edge_agg<<<aggGrid, 256, 0, stream>>>(bufA, bufB, ei, ei + E, E);
    gemm_bias_relu<128,128><<<gemmGridN, 128, 0, stream>>>(bufB, w1b, b1b, bufA, Nn);
    gemm_bias_relu<128,128><<<gemmGridN, 128, 0, stream>>>(bufA, w2b, b2b, bufB, Nn);

    // ---- row-wise L2 normalize: bufA = normalize(bufB) ----
    l2norm_rows<<<dim3((unsigned)(((long long)Nn * 32 + 255) / 256)), 256, 0, stream>>>(
        bufB, bufA, Nn);

    // ---- SOPOOL per-node MLP (s buffers live inside bufB region) ----
    float* s1 = bufB;
    float* s2 = bufB + (size_t)Nn * 32;
    float* s3 = bufB + (size_t)Nn * 64;
    gemm_bias_relu<128,32><<<gemmGridN, 128, 0, stream>>>(bufA, ws1, bs1, s1, Nn);
    gemm_bias_relu<32, 32><<<gemmGridN, 128, 0, stream>>>(s1,  ws2, bs2, s2, Nn);
    gemm_bias_relu<32, 32><<<gemmGridN, 128, 0, stream>>>(s2,  ws3, bs3, s3, Nn);

    // ---- second-order pooling -> HH (first G*1024 floats of d_out) ----
    float* HH = (float*)d_out;
    sopool<<<dim3((unsigned)G), 128, 0, stream>>>(s3, HH);

    // ---- graph MLP: 1024 -> 32 -> 32 -> 2 ----
    float* o1 = bufA;                     // hn no longer needed
    float* o2 = bufA + (size_t)G * 32;
    gemm_bias_relu<1024,32><<<gemmGridG, 128, 0, stream>>>(HH, wm1, bm1, o1, G);
    gemm_bias_relu<32,  32><<<gemmGridG, 128, 0, stream>>>(o1, wm2, bm2, o2, G);
    final_fc<<<dim3((unsigned)((G * 2 + 255) / 256)), 256, 0, stream>>>(
        o2, Wm3, bm3, HH + (size_t)G * 1024, G);
}
